// EGATConvGNN_22711787061918
// MI455X (gfx1250) — matrix operand order repaired and checked by reading, not compiled, not run on hardware
//
#include <hip/hip_runtime.h>

#define NN 50000
#define NE 800000

typedef __attribute__((ext_vector_type(8)))  float  v8f_t;
typedef __attribute__((ext_vector_type(16))) __bf16 v16bf_t;

union egat_bfvec { unsigned short u[16]; uint4 q[2]; v16bf_t v; };

__device__ __forceinline__ float egat_lrelu(float x) { return x >= 0.0f ? x : 0.01f * x; }

__device__ __forceinline__ unsigned short egat_f2bf(float f) {
    unsigned u = __float_as_uint(f);
    u += 0x7FFFu + ((u >> 16) & 1u);          // round-to-nearest-even
    return (unsigned short)(u >> 16);
}
// order-preserving f32 -> u32 encoding for atomic max (0 encodes "below -inf")
__device__ __forceinline__ unsigned egat_ord(float f) {
    unsigned u = __float_as_uint(f);
    return (u & 0x80000000u) ? ~u : (u | 0x80000000u);
}
__device__ __forceinline__ float egat_unord(unsigned e) {
    return (e & 0x80000000u) ? __uint_as_float(e ^ 0x80000000u) : __uint_as_float(~e);
}

// ---------------------------------------------------------------------------
// Pack f32 weight W[K x N] into WMMA B-fragment layout, bf16, zero-padded in K.
// Tile (kt, nt): 32 lanes x 16 bf16 contiguous (32B per lane).
// lane L: col = nt*16 + (L%16); element j <-> K = kt*32 + 8*(L/16) + j (j<8), kt*32 + 16 + 8*(L/16) + (j-8) (j>=8).
// ---------------------------------------------------------------------------
__global__ void egat_pack_w(const float* __restrict__ W, unsigned short* __restrict__ out,
                            int K, int N) {
    int KT = (K + 31) / 32;
    int total = KT * (N / 16) * 32;
    int tid = blockIdx.x * blockDim.x + threadIdx.x;
    if (tid >= total) return;
    int lane = tid & 31;
    int tile = tid >> 5;
    int nt = tile % (N / 16);
    int kt = tile / (N / 16);
    int half = lane >> 4;
    int n = nt * 16 + (lane & 15);
    egat_bfvec frag;
#pragma unroll
    for (int j = 0; j < 16; ++j) {
        int k = kt * 32 + half * 8 + (j < 8 ? j : 8 + j);
        frag.u[j] = (k < K) ? egat_f2bf(W[(size_t)k * N + n]) : (unsigned short)0;
    }
    uint4* o = (uint4*)(out + (size_t)tid * 16);
    o[0] = frag.q[0];
    o[1] = frag.q[1];
}

// ---------------------------------------------------------------------------
// WMMA GEMM: C[M x N](f32) = A[M x K](bf16 row-major) @ Bp(packed bf16).
// K compile-time multiple of 32. One wave computes a 16x64 C strip
// (4 N-tiles), v_wmma_f32_16x16x32_bf16, fully unrolled k loop.
// Grid: (N/64, M/16), block 32.
// ---------------------------------------------------------------------------
template<int K>
__global__ void egat_gemm_wmma(const unsigned short* __restrict__ Abf,
                               const unsigned short* __restrict__ Bp,
                               float* __restrict__ C, int N) {
    constexpr int KT = K / 32;
    const int lane = threadIdx.x & 31;
    const int half = lane >> 4;
    const int r    = lane & 15;
    const int m0   = blockIdx.y * 16;
    const int nt0  = blockIdx.x * 4;       // first of 4 n-tiles
    const int NT16 = N >> 4;

    v8f_t acc[4];
#pragma unroll
    for (int t = 0; t < 4; ++t)
#pragma unroll
        for (int i = 0; i < 8; ++i) acc[t][i] = 0.0f;

    // A row for this lane; +8*half folds the half-offset of the fragment map
    const unsigned short* arow = Abf + (size_t)(m0 + r) * K + 8 * half;

#pragma unroll
    for (int kt = 0; kt < KT; ++kt) {
        egat_bfvec a;
        a.q[0] = *(const uint4*)(arow + kt * 32);        // elements 0..7
        a.q[1] = *(const uint4*)(arow + kt * 32 + 16);   // elements 8..15
#pragma unroll
        for (int nt = 0; nt < 4; ++nt) {
            egat_bfvec b;
            const uint4* bp = (const uint4*)(Bp +
                (((size_t)kt * NT16 + (nt0 + nt)) * 32 + lane) * 16);
            b.q[0] = bp[0];
            b.q[1] = bp[1];
            acc[nt] = __builtin_amdgcn_wmma_f32_16x16x32_bf16(false, a.v, false, b.v,
                                                              (short)0, acc[nt], false, false);
        }
    }
    // C element i <-> row m0 + i + 8*half, col nt*16 + r
#pragma unroll
    for (int nt = 0; nt < 4; ++nt) {
        float* crow = C + (size_t)(m0 + (half << 3)) * N + (nt0 + nt) * 16 + r;
#pragma unroll
        for (int i = 0; i < 8; ++i) crow[(size_t)i * N] = acc[nt][i];
    }
}

// ---------------------------------------------------------------------------
// Elementwise / graph kernels
// ---------------------------------------------------------------------------
__global__ void egat_zero_u32(unsigned* __restrict__ p, size_t n) {
    size_t i = (size_t)blockIdx.x * blockDim.x + threadIdx.x;
    if (i < n) p[i] = 0u;
}

// h0 = lrelu(h_in @ Wh + bh), emitted as f32 (for graph math) + bf16 (GEMM A)
__global__ void egat_proj_h(const float* __restrict__ hin, const float* __restrict__ W,
                            const float* __restrict__ b, float* __restrict__ hout,
                            unsigned short* __restrict__ hbf) {
    int idx = blockIdx.x * blockDim.x + threadIdx.x;
    if (idx >= NN * 32) return;
    int n = idx >> 5, c = idx & 31;
    float s = b[c];
#pragma unroll
    for (int i = 0; i < 6; ++i) s += hin[n * 6 + i] * W[i * 32 + c];
    s = egat_lrelu(s);
    hout[idx] = s;
    hbf[idx]  = egat_f2bf(s);
}

// e0 = e_in @ We + be, written directly as bf16 zero-padded 8 -> 32 cols
// (the f32 edge tensor is only ever consumed as a GEMM input)
__global__ void egat_proj_e(const float* __restrict__ ein, const float* __restrict__ W,
                            const float* __restrict__ b, unsigned short* __restrict__ ebf) {
    size_t idx = (size_t)blockIdx.x * blockDim.x + threadIdx.x;
    if (idx >= (size_t)NE * 32) return;
    size_t i = idx >> 5; int c = (int)(idx & 31);
    unsigned short o = 0;
    if (c < 8) o = egat_f2bf(ein[i] * W[c] + b[c]);
    ebf[idx] = o;
}

// e_out holds f_fij on entry; f_out = lrelu(f_ni[src] + f_nj[dst] + f_fij + bias).
// Also emits bf16 copy (next layer's GEMM A); ebf==nullptr on the last layer.
__global__ void egat_edge_combine(const float* __restrict__ f_ni, const float* __restrict__ f_nj,
                                  const float* __restrict__ bias, const int* __restrict__ src,
                                  const int* __restrict__ dst, float* __restrict__ e_out,
                                  unsigned short* __restrict__ ebf) {
    size_t idx = (size_t)blockIdx.x * blockDim.x + threadIdx.x;
    if (idx >= (size_t)NE * 64) return;
    size_t i = idx >> 6; int c = (int)(idx & 63);
    float v = f_ni[(size_t)src[i] * 64 + c] + f_nj[(size_t)dst[i] * 64 + c]
            + e_out[idx] + bias[c];
    v = egat_lrelu(v);
    e_out[idx] = v;
    if (ebf) ebf[idx] = egat_f2bf(v);
}

// scores[e,h] = sum_d f_out[e,h,d] * attn[h,d]; running segment-max via atomics
__global__ void egat_edge_scores(const float* __restrict__ e_f, const float* __restrict__ attn,
                                 const int* __restrict__ dst, float* __restrict__ scores,
                                 unsigned* __restrict__ smax, int hds) {
    int tid = blockIdx.x * blockDim.x + threadIdx.x;
    if (tid >= NE * hds) return;
    int i = tid / hds, h = tid - i * hds;
    int de = 64 / hds;
    const float* p = e_f + (size_t)i * 64 + h * de;
    const float* a = attn + h * de;
    float s = 0.0f;
    for (int d = 0; d < de; d += 4) {
        float4 x = *(const float4*)(p + d);
        float4 w = *(const float4*)(a + d);
        s += x.x * w.x + x.y * w.y + x.z * w.z + x.w * w.w;
    }
    scores[tid] = s;
    atomicMax(&smax[(size_t)dst[i] * hds + h], egat_ord(s));
}

__global__ void egat_edge_expsum(const int* __restrict__ dst, float* __restrict__ scores,
                                 const unsigned* __restrict__ smax, float* __restrict__ ssum,
                                 int hds) {
    int tid = blockIdx.x * blockDim.x + threadIdx.x;
    if (tid >= NE * hds) return;
    int i = tid / hds, h = tid - i * hds;
    float mx = egat_unord(smax[(size_t)dst[i] * hds + h]);
    float ex = __expf(scores[tid] - mx);
    scores[tid] = ex;
    atomicAdd(&ssum[(size_t)dst[i] * hds + h], ex);
}

// h_acc[dst, c] += h_src[src, c] * (ex / ssum[dst, head(c)])
__global__ void egat_edge_scatter(const float* __restrict__ h_src, const float* __restrict__ scores,
                                  const float* __restrict__ ssum, const int* __restrict__ src,
                                  const int* __restrict__ dst, float* __restrict__ h_acc,
                                  int hds, int outn) {
    size_t idx = (size_t)blockIdx.x * blockDim.x + threadIdx.x;
    if (idx >= (size_t)NE * 256) return;
    size_t i = idx >> 8; int c = (int)(idx & 255);
    int hd = c / outn;
    int d = dst[i];
    float alpha = scores[i * hds + hd] / ssum[(size_t)d * hds + hd];
    atomicAdd(&h_acc[(size_t)d * 256 + c], h_src[(size_t)src[i] * 256 + c] * alpha);
}

// h = lrelu(h_acc), emitted as f32 + bf16 (next layer's GEMM A)
__global__ void egat_lrelu_cvt(const float* __restrict__ in, float* __restrict__ out,
                               unsigned short* __restrict__ obf, size_t n) {
    size_t i = (size_t)blockIdx.x * blockDim.x + threadIdx.x;
    if (i >= n) return;
    float v = egat_lrelu(in[i]);
    out[i] = v;
    obf[i] = egat_f2bf(v);
}

// column-sum partials over 256-wide rows; coalesced (thread t owns column t)
__global__ void egat_col_mean(const float* __restrict__ h, float* __restrict__ mean, int rpb) {
    int c = threadIdx.x;
    int r0 = blockIdx.x * rpb;
    int r1 = r0 + rpb; if (r1 > NN) r1 = NN;
    float s = 0.0f;
    for (int r = r0; r < r1; ++r) s += h[(size_t)r * 256 + c];
    atomicAdd(&mean[c], s);
}

__global__ void egat_pred_head(const float* __restrict__ mean,
                               const float* __restrict__ p1W, const float* __restrict__ p1b,
                               const float* __restrict__ p2W, const float* __restrict__ p2b,
                               const float* __restrict__ r1W, const float* __restrict__ r1b,
                               const float* __restrict__ r2W, const float* __restrict__ r2b,
                               float* __restrict__ out) {
    __shared__ float hm[256], x1[64], x2[32];
    int t = threadIdx.x;
    hm[t] = mean[t] * (1.0f / (float)NN);
    __syncthreads();
    if (t < 64) {
        float s = p1b[t];
        for (int i = 0; i < 256; ++i) s += hm[i] * p1W[i * 64 + t];
        x1[t] = egat_lrelu(s);
    }
    __syncthreads();
    if (t < 32) {
        float s = p2b[t];
        for (int i = 0; i < 64; ++i) s += x1[i] * p2W[i * 32 + t];
        x2[t] = egat_lrelu(s);
    }
    __syncthreads();
    if (t < 4) {
        int j = t & 1;
        const float* W = (t < 2) ? r1W : r2W;
        const float* b = (t < 2) ? r1b : r2b;
        float s = b[j];
        for (int i = 0; i < 32; ++i) s += x2[i] * W[i * 2 + j];
        out[t] = 1.0f / (1.0f + __expf(-s));
    }
}

// ---------------------------------------------------------------------------
extern "C" void kernel_launch(void* const* d_in, const int* in_sizes, int n_in,
                              void* d_out, int out_size, void* d_ws, size_t ws_size,
                              hipStream_t stream) {
    (void)in_sizes; (void)n_in; (void)out_size; (void)ws_size;

    const float* h_in   = (const float*)d_in[0];
    const float* e_in   = (const float*)d_in[1];
    const int*   src    = (const int*)  d_in[2];
    const int*   dst    = (const int*)  d_in[3];
    const float* phW    = (const float*)d_in[4];
    const float* phB    = (const float*)d_in[5];
    const float* peW    = (const float*)d_in[6];
    const float* peB    = (const float*)d_in[7];
    float* out = (float*)d_out;

    // workspace bump allocator (256B aligned)
    char* base = (char*)d_ws;
    size_t off = 0;
    auto alloc = [&](size_t bytes) -> void* {
        off = (off + 255) & ~(size_t)255;
        void* p = (void*)(base + off);
        off += bytes;
        return p;
    };
    float*    hA     = (float*)alloc((size_t)NN * 256 * 4);   // node features (f32)
    float*    hB     = (float*)alloc((size_t)NN * 256 * 4);   // node accumulator
    float*    eF     = (float*)alloc((size_t)NE * 64 * 4);    // edge f32 (intra-layer only)
    float*    fni    = (float*)alloc((size_t)NN * 64 * 4);
    float*    fnj    = (float*)alloc((size_t)NN * 64 * 4);
    float*    hsrc   = (float*)alloc((size_t)NN * 256 * 4);
    float*    scores = (float*)alloc((size_t)NE * 4 * 4);
    unsigned* smax   = (unsigned*)alloc((size_t)NN * 4 * 4);
    float*    ssum   = (float*)alloc((size_t)NN * 4 * 4);
    float*    meanb  = (float*)alloc(256 * 4);
    unsigned short* hAbf = (unsigned short*)alloc((size_t)NN * 256 * 2);
    unsigned short* eBf  = (unsigned short*)alloc((size_t)NE * 64 * 2);
    unsigned short* WniP   = (unsigned short*)alloc(256 * 64 * 2);
    unsigned short* WnjP   = (unsigned short*)alloc(256 * 64 * 2);
    unsigned short* WfijP  = (unsigned short*)alloc(64 * 64 * 2);
    unsigned short* WnodeP = (unsigned short*)alloc(256 * 256 * 2);

    const int TB = 256;
    auto blocks = [](size_t n, int tb) { return (unsigned)((n + tb - 1) / tb); };

    auto launch_gemm = [&](int K, const unsigned short* A, const unsigned short* Bp,
                           float* C, int M, int N) {
        dim3 g(N / 64, M / 16);
        if (K == 32)
            egat_gemm_wmma<32><<<g, 32, 0, stream>>>(A, Bp, C, N);
        else if (K == 64)
            egat_gemm_wmma<64><<<g, 32, 0, stream>>>(A, Bp, C, N);
        else
            egat_gemm_wmma<256><<<g, 32, 0, stream>>>(A, Bp, C, N);
    };

    // ---- input projections (emit bf16 GEMM inputs directly) ----
    egat_proj_h<<<blocks((size_t)NN * 32, TB), TB, 0, stream>>>(h_in, phW, phB, hA, hAbf);
    egat_proj_e<<<blocks((size_t)NE * 32, TB), TB, 0, stream>>>(e_in, peW, peB, eBf);

    const int inn_[4]  = {32, 256, 256, 256};
    const int ine_[4]  = {8, 64, 64, 64};
    const int kpe_[4]  = {32, 64, 64, 64};     // edge K padded to x32
    const int hds_[4]  = {4, 4, 4, 1};
    const int outn_[4] = {64, 64, 64, 256};

    for (int L = 0; L < 4; ++L) {
        const float* Wni   = (const float*)d_in[8 + 6 * L + 0];
        const float* Wnj   = (const float*)d_in[8 + 6 * L + 1];
        const float* Wfij  = (const float*)d_in[8 + 6 * L + 2];
        const float* Wnode = (const float*)d_in[8 + 6 * L + 3];
        const float* attn  = (const float*)d_in[8 + 6 * L + 4];
        const float* biasL = (const float*)d_in[8 + 6 * L + 5];
        const int K    = inn_[L];
        const int Ke   = ine_[L];
        const int Kep  = kpe_[L];
        const int hds  = hds_[L];
        const int outn = outn_[L];

        // zero accumulators: hB (N*256), smax (encoded -inf == 0), ssum
        egat_zero_u32<<<blocks((size_t)NN * 256, TB), TB, 0, stream>>>((unsigned*)hB, (size_t)NN * 256);
        egat_zero_u32<<<blocks((size_t)NN * hds, TB), TB, 0, stream>>>(smax, (size_t)NN * hds);
        egat_zero_u32<<<blocks((size_t)NN * hds, TB), TB, 0, stream>>>((unsigned*)ssum, (size_t)NN * hds);

        // pack weights into WMMA B-fragment layout (zero-pad K)
        egat_pack_w<<<blocks((size_t)((K + 31) / 32) * (64 / 16) * 32, TB), TB, 0, stream>>>(Wni, WniP, K, 64);
        egat_pack_w<<<blocks((size_t)((K + 31) / 32) * (64 / 16) * 32, TB), TB, 0, stream>>>(Wnj, WnjP, K, 64);
        egat_pack_w<<<blocks((size_t)((Ke + 31) / 32) * (64 / 16) * 32, TB), TB, 0, stream>>>(Wfij, WfijP, Ke, 64);
        egat_pack_w<<<blocks((size_t)((K + 31) / 32) * (256 / 16) * 32, TB), TB, 0, stream>>>(Wnode, WnodeP, K, 256);

        // GEMMs (WMMA): f_ni, f_nj (N x 64), h_src (N x 256), f_fij -> eF (E x 64)
        launch_gemm(K, hAbf, WniP, fni, NN, 64);
        launch_gemm(K, hAbf, WnjP, fnj, NN, 64);
        launch_gemm(K, hAbf, WnodeP, hsrc, NN, 256);
        launch_gemm(Kep, eBf, WfijP, eF, NE, 64);

        // graph passes (edge bf16 for next layer fused into combine; none on last)
        egat_edge_combine<<<blocks((size_t)NE * 64, TB), TB, 0, stream>>>(
            fni, fnj, biasL, src, dst, eF, (L < 3) ? eBf : (unsigned short*)nullptr);
        egat_edge_scores<<<blocks((size_t)NE * hds, TB), TB, 0, stream>>>(eF, attn, dst, scores, smax, hds);
        egat_edge_expsum<<<blocks((size_t)NE * hds, TB), TB, 0, stream>>>(dst, scores, smax, ssum, hds);
        egat_edge_scatter<<<blocks((size_t)NE * 256, TB), TB, 0, stream>>>(hsrc, scores, ssum, src, dst, hB, hds, outn);

        // h = lrelu(h_out) -> f32 + bf16
        egat_lrelu_cvt<<<blocks((size_t)NN * 256, TB), TB, 0, stream>>>(hB, hA, hAbf, (size_t)NN * 256);
    }

    // ---- prediction head ----
    const float* p1W = (const float*)d_in[32]; const float* p1b = (const float*)d_in[33];
    const float* p2W = (const float*)d_in[34]; const float* p2b = (const float*)d_in[35];
    const float* r1W = (const float*)d_in[36]; const float* r1b = (const float*)d_in[37];
    const float* r2W = (const float*)d_in[38]; const float* r2b = (const float*)d_in[39];

    egat_zero_u32<<<1, 256, 0, stream>>>((unsigned*)meanb, 256);
    egat_col_mean<<<256, 256, 0, stream>>>(hA, meanb, (NN + 255) / 256);
    egat_pred_head<<<1, 256, 0, stream>>>(meanb, p1W, p1b, p2W, p2b, r1W, r1b, r2W, r2b, out);
}
